// Quantize_onehot_VQVAE_22892175687685
// MI455X (gfx1250) — compile-verified, matrix operand order checked by reading, not run on hardware
//
#include <hip/hip_runtime.h>
#include <cstdint>

typedef __attribute__((ext_vector_type(16))) __bf16 v16bf;
typedef __attribute__((ext_vector_type(8)))  float  v8f;

// Problem constants
#define ROWS   32768      // B*N*G
#define DK     256        // d
#define NE     2048
#define TAU_INV 2.0f      // 1/0.5

static __device__ __forceinline__ uint32_t pack_bf16x2(float a, float b) {
  union { __bf16 h[2]; uint32_t u; } x;
  x.h[0] = (__bf16)a; x.h[1] = (__bf16)b;
  return x.u;
}

// ---------------- init row-max keys ----------------
__global__ void vq_init(unsigned long long* __restrict__ rowkey) {
  int i = blockIdx.x * blockDim.x + threadIdx.x;
  if (i < ROWS) rowkey[i] = 0ULL;
}

// ---------------- row squared-norms (len 256), one wave per row ----------------
__global__ __launch_bounds__(256) void row_norm_256(
    const float* __restrict__ m, float* __restrict__ out, int rows) {
  const int t = threadIdx.x, lane = t & 31;
  const int row = blockIdx.x * 8 + (t >> 5);
  if (row >= rows) return;
  const float* p = m + (size_t)row * DK + lane * 8;
  float4 a = *reinterpret_cast<const float4*>(p);
  float4 b = *reinterpret_cast<const float4*>(p + 4);
  float s = a.x*a.x + a.y*a.y + a.z*a.z + a.w*a.w
          + b.x*b.x + b.y*b.y + b.z*b.z + b.w*b.w;
#pragma unroll
  for (int off = 16; off > 0; off >>= 1) s += __shfl_xor(s, off, 32);
  if (lane == 0) out[row] = s;
}

// ---------------- fused bf16-WMMA GEMM + logits + row argmax ----------------
__global__ __launch_bounds__(256) void vq_gemm_logits(
    const float* __restrict__ z, const float* __restrict__ emb,
    const float* __restrict__ gum, const float* __restrict__ xnorm,
    const float* __restrict__ enorm, unsigned long long* __restrict__ rowkey,
    float* __restrict__ logits) {
  constexpr int LDK = 72;               // 64 + 8 pad halves -> 144B rows, bank-clean
  __shared__ __bf16 sA[128 * LDK];
  __shared__ __bf16 sB[128 * LDK];
  const int t = threadIdx.x;
  const int lane = t & 31;
  const int wave = t >> 5;
  const int wM = wave >> 1, wN = wave & 1;   // 4 x 2 wave grid
  const int bN = blockIdx.x, bM = blockIdx.y;

  v8f acc[2][4] = {};

  for (int kc = 0; kc < 4; ++kc) {           // K chunks of 64
#pragma unroll
    for (int it = 0; it < 8; ++it) {         // 128x64 fp32 -> bf16 into LDS
      const int fid = it * 256 + t;          // float4 id in [0,2048)
      const int row = fid >> 4;              // 16 float4 per 64-float row
      const int c4  = fid & 15;
      const float4 a = *reinterpret_cast<const float4*>(
          z + ((size_t)(bM * 128 + row)) * DK + kc * 64 + c4 * 4);
      uint2 pa; pa.x = pack_bf16x2(a.x, a.y); pa.y = pack_bf16x2(a.z, a.w);
      *reinterpret_cast<uint2*>(&sA[row * LDK + c4 * 4]) = pa;
      const float4 b = *reinterpret_cast<const float4*>(
          emb + ((size_t)(bN * 128 + row)) * DK + kc * 64 + c4 * 4);
      uint2 pb; pb.x = pack_bf16x2(b.x, b.y); pb.y = pack_bf16x2(b.z, b.w);
      *reinterpret_cast<uint2*>(&sB[row * LDK + c4 * 4]) = pb;
    }
    __syncthreads();
#pragma unroll
    for (int ks = 0; ks < 2; ++ks) {         // two K=32 WMMA steps per chunk
      const int kb = ks * 32 + ((lane < 16) ? 0 : 8);
      union Frag { uint4 q[2]; v16bf v; };
      Frag fa[2], fb[4];
#pragma unroll
      for (int mt = 0; mt < 2; ++mt) {
        const int r = wM * 32 + mt * 16 + (lane & 15);
        fa[mt].q[0] = *reinterpret_cast<const uint4*>(&sA[r * LDK + kb]);
        fa[mt].q[1] = *reinterpret_cast<const uint4*>(&sA[r * LDK + kb + 16]);
      }
#pragma unroll
      for (int nt = 0; nt < 4; ++nt) {
        const int r = wN * 64 + nt * 16 + (lane & 15);
        fb[nt].q[0] = *reinterpret_cast<const uint4*>(&sB[r * LDK + kb]);
        fb[nt].q[1] = *reinterpret_cast<const uint4*>(&sB[r * LDK + kb + 16]);
      }
#pragma unroll
      for (int mt = 0; mt < 2; ++mt)
#pragma unroll
        for (int nt = 0; nt < 4; ++nt)
          acc[mt][nt] = __builtin_amdgcn_wmma_f32_16x16x32_bf16(
              false, fa[mt].v, false, fb[nt].v, (short)0, acc[mt][nt],
              false, false);
    }
    __syncthreads();
  }

  // epilogue: logit = ( -(||x||^2 - 2x.e + ||e||^2) + gumbel ) / tau
  float en[4];
#pragma unroll
  for (int nt = 0; nt < 4; ++nt)
    en[nt] = enorm[bN * 128 + wN * 64 + nt * 16 + (lane & 15)];
  const int mhi = (lane & 16) ? 8 : 0;       // C/D layout: lanes 16-31 hold M+8

#pragma unroll
  for (int mt = 0; mt < 2; ++mt) {
    const int ibase = bM * 128 + wM * 32 + mt * 16 + mhi;
#pragma unroll
    for (int r = 0; r < 8; ++r) {
      const int i = ibase + r;
      const float xn = xnorm[i];
      unsigned long long key = 0ULL;
#pragma unroll
      for (int nt = 0; nt < 4; ++nt) {
        const int j = bN * 128 + wN * 64 + nt * 16 + (lane & 15);
        const float dot = acc[mt][nt][r];
        const float lg =
            (2.0f * dot - xn - en[nt] + gum[(size_t)i * NE + j]) * TAU_INV;
        logits[(size_t)i * NE + j] = lg;
        uint32_t ub = __float_as_uint(lg);          // monotone float -> uint
        ub = (ub & 0x80000000u) ? ~ub : (ub | 0x80000000u);
        const unsigned long long k2 =
            ((unsigned long long)ub << 32) | (uint32_t)(~(uint32_t)j);
        key = (k2 > key) ? k2 : key;                // ~j: first max wins ties
      }
#pragma unroll
      for (int off = 1; off < 16; off <<= 1) {      // reduce over the 16 N-lanes
        uint32_t lo = (uint32_t)key, hi = (uint32_t)(key >> 32);
        lo = __shfl_xor(lo, off, 32);
        hi = __shfl_xor(hi, off, 32);
        const unsigned long long o = ((unsigned long long)hi << 32) | lo;
        key = (o > key) ? o : key;
      }
      if ((lane & 15) == 0) atomicMax(&rowkey[i], key);
    }
  }
}

// ---------------- per-row softmax + straight-through output + loss partial ----
__global__ __launch_bounds__(256) void vq_softmax_out(
    const float* __restrict__ z, const float* __restrict__ emb,
    const unsigned long long* __restrict__ rowkey,
    float* __restrict__ logits /* in-place -> onehot */,
    float* __restrict__ ind_out, float* __restrict__ rowsum) {
  __shared__ float2 red[256];
  const int i = blockIdx.x;
  const int t = threadIdx.x;
  const unsigned long long key = rowkey[i];
  const int ind = (int)(~(uint32_t)key);
  const uint32_t um = (uint32_t)(key >> 32);
  const uint32_t mb = (um & 0x80000000u) ? (um & 0x7FFFFFFFu) : ~um;
  const float mx = __uint_as_float(mb);

  float* row = logits + (size_t)i * NE;
  float4 a = *reinterpret_cast<const float4*>(row + t * 8);
  float4 b = *reinterpret_cast<const float4*>(row + t * 8 + 4);
  float e[8];
  e[0] = __expf(a.x - mx); e[1] = __expf(a.y - mx);
  e[2] = __expf(a.z - mx); e[3] = __expf(a.w - mx);
  e[4] = __expf(b.x - mx); e[5] = __expf(b.y - mx);
  e[6] = __expf(b.z - mx); e[7] = __expf(b.w - mx);
  float se = ((e[0] + e[1]) + (e[2] + e[3])) + ((e[4] + e[5]) + (e[6] + e[7]));

  const float dz = emb[(size_t)ind * DK + t] - z[(size_t)i * DK + t];

  red[t].x = se; red[t].y = dz * dz;
  __syncthreads();
  for (int s = 128; s > 0; s >>= 1) {
    if (t < s) { red[t].x += red[t + s].x; red[t].y += red[t + s].y; }
    __syncthreads();
  }
  const float rinv = 1.0f / red[0].x;
#pragma unroll
  for (int k = 0; k < 8; ++k) {
    const int j = t * 8 + k;
    const float p = e[k] * rinv;
    const float hard = (j == ind) ? 1.0f : 0.0f;
    e[k] = (hard - p) + p;                 // straight-through, reference rounding
  }
  float4 oa{e[0], e[1], e[2], e[3]}, ob{e[4], e[5], e[6], e[7]};
  *reinterpret_cast<float4*>(row + t * 8) = oa;
  *reinterpret_cast<float4*>(row + t * 8 + 4) = ob;
  if (t == 0) { rowsum[i] = red[0].y; ind_out[i] = (float)ind; }
}

// ---------------- final deterministic loss reduction ----------------
__global__ __launch_bounds__(256) void vq_diff(
    const float* __restrict__ rowsum, float* __restrict__ diff_out) {
  __shared__ float red[256];
  const int t = threadIdx.x;
  float s = 0.0f;
  for (int k = t; k < ROWS; k += 256) s += rowsum[k];
  red[t] = s;
  __syncthreads();
  for (int st = 128; st > 0; st >>= 1) {
    if (t < st) red[t] += red[t + st];
    __syncthreads();
  }
  // 10*(0.25*m + m) = 12.5*m,  m = sum / (32768*256)
  if (t == 0) diff_out[0] = 12.5f * red[0] * (1.0f / 8388608.0f);
}

extern "C" void kernel_launch(void* const* d_in, const int* in_sizes, int n_in,
                              void* d_out, int out_size, void* d_ws, size_t ws_size,
                              hipStream_t stream) {
  const float* z   = (const float*)d_in[0];   // (8,1024,1024) == (32768,256) flat
  const float* emb = (const float*)d_in[1];   // (2048,256)
  const float* gum = (const float*)d_in[2];   // (32768,2048)

  float* out    = (float*)d_out;
  float* onehot = out;                         // 67,108,864 floats
  float* diff   = out + (size_t)ROWS * NE;     // 1 float
  float* indf   = diff + 1;                    // 32,768 floats

  char* ws = (char*)d_ws;
  float* xnorm = (float*)(ws);                                   // 32768 f
  float* enorm = (float*)(ws + 131072);                          // 2048 f
  unsigned long long* rowkey = (unsigned long long*)(ws + 139264); // 32768 u64
  float* rowsum = (float*)(ws + 401408);                         // 32768 f

  vq_init<<<ROWS / 256, 256, 0, stream>>>(rowkey);
  row_norm_256<<<ROWS / 8, 256, 0, stream>>>(z, xnorm, ROWS);
  row_norm_256<<<NE / 8, 256, 0, stream>>>(emb, enorm, NE);
  vq_gemm_logits<<<dim3(NE / 128, ROWS / 128), 256, 0, stream>>>(
      z, emb, gum, xnorm, enorm, rowkey, onehot);
  vq_softmax_out<<<ROWS, 256, 0, stream>>>(z, emb, rowkey, onehot, indf, rowsum);
  vq_diff<<<1, 256, 0, stream>>>(rowsum, diff);
}
